// GCMCModel_50302656971285
// MI455X (gfx1250) — matrix operand order repaired and checked by reading, not compiled, not run on hardware
//
#include <hip/hip_runtime.h>
#include <hip/hip_bf16.h>

typedef __attribute__((ext_vector_type(16))) _Float16 v16h;
typedef __attribute__((ext_vector_type(8)))  _Float16 v8h;
typedef __attribute__((ext_vector_type(4)))  _Float16 v4h;
typedef __attribute__((ext_vector_type(8)))  float    v8f;

#define N_USER 100000
#define N_ITEM 50000
#define DIM    64
#define BATCH  16384
#define NEDGE  2000000

// ---------------------------------------------------------------------------
// WMMA helpers (gfx1250 wave32 layouts, per CDNA5 ISA §7.12.2)
// ---------------------------------------------------------------------------
__device__ __forceinline__ v8f wmma16(v16h a, v16h b, v8f c) {
  // D = A(16x32 f16) x B(32x16 f16) + C(16x16 f32)
  return __builtin_amdgcn_wmma_f32_16x16x32_f16(
      /*neg_a=*/false, a, /*neg_b=*/false, b,
      /*c_mod=*/(short)0, c, /*reuse_a=*/false, /*reuse_b=*/false);
}

__device__ __forceinline__ v16h cat8(v8h lo, v8h hi) {
  return __builtin_shufflevector(lo, hi, 0, 1, 2, 3, 4, 5, 6, 7,
                                         8, 9, 10, 11, 12, 13, 14, 15);
}

// A fragment from row-major f16 [16][ld] in LDS, K-window [k0, k0+32).
// lane<16: M=lane, elems 0..7 = K k0..k0+7, elems 8..15 = K k0+16..k0+23
// lane>=16: M=lane-16, elems 0..7 = K k0+8..+15, elems 8..15 = K k0+24..+31
__device__ __forceinline__ v16h load_a_lds(const _Float16* base, int ld, int k0, int lane) {
  const int m  = lane & 15;
  const int kb = k0 + ((lane & 16) ? 8 : 0);
  const _Float16* p = base + m * ld + kb;            // 16B aligned (kb mult of 8)
  v8h lo = *(const v8h*)(p);
  v8h hi = *(const v8h*)(p + 16);
  return cat8(lo, hi);
}

// A fragment formed as packed elementwise product of two f16 [16][64] LDS
// buffers (K-slice of the concatenated-product feature matrix, on the fly).
__device__ __forceinline__ v16h load_a_prod(const _Float16* X, const _Float16* Y,
                                            int k0, int lane) {
  const int m  = lane & 15;
  const int kb = k0 + ((lane & 16) ? 8 : 0);
  const _Float16* px = X + m * 64 + kb;
  const _Float16* py = Y + m * 64 + kb;
  v8h x0 = *(const v8h*)(px);
  v8h x1 = *(const v8h*)(px + 16);
  v8h y0 = *(const v8h*)(py);
  v8h y1 = *(const v8h*)(py + 16);
  return cat8(x0 * y0, x1 * y1);                     // v_pk_mul_f16
}

// B fragment from fragment-major packed weights: each lane's 16 halves are
// contiguous and 32B aligned -> two global_load_b128, no packing movs.
__device__ __forceinline__ v16h load_b_packed(const _Float16* __restrict__ P,
                                              int frag, int lane) {
  return *(const v16h*)(P + ((size_t)frag * 32 + lane) * 16);
}

// GCN layer: sA holds 16x64 f16 hidden; computes relu(sA @ W + bias) and
// overwrites sA with the result (A fragments held in registers first).
// Packed W fragment order: frag = nt*2 + kt (KT=2, NT=4).
__device__ __forceinline__ void gcn_layer(_Float16* sA, const _Float16* __restrict__ P,
                                          const float* __restrict__ bias, int lane) {
  v16h a0 = load_a_lds(sA, 64, 0, lane);
  v16h a1 = load_a_lds(sA, 64, 32, lane);
  const int coln  = lane & 15;
  const int rbase = (lane & 16) ? 8 : 0;
#pragma unroll
  for (int nt = 0; nt < 4; ++nt) {
    v8f c = {};
    c = wmma16(a0, load_b_packed(P, nt * 2 + 0, lane), c);
    c = wmma16(a1, load_b_packed(P, nt * 2 + 1, lane), c);
    const int col = nt * 16 + coln;
    const float bv = bias[col];
#pragma unroll
    for (int j = 0; j < 8; ++j) {
      float v = c[j] + bv;
      sA[(rbase + j) * 64 + col] = (_Float16)(v > 0.f ? v : 0.f);
    }
  }
}

// ---------------------------------------------------------------------------
// Stage kernels
// ---------------------------------------------------------------------------
__global__ void zero_ws(float4* __restrict__ p, int n4) {
  const int stride = gridDim.x * blockDim.x;
  for (int i = blockIdx.x * blockDim.x + threadIdx.x; i < n4; i += stride)
    p[i] = make_float4(0.f, 0.f, 0.f, 0.f);
}

// Convert f32 weight [K][N] into f16 fragment-major layout:
//   P[((nt*KT + kt)*32 + lane)*16 + e] = W[kt*32 + ((lane&16)?8:0) + (e<8 ? e : e+8)][nt*16 + (lane&15)]
__global__ void pack_w(const float* __restrict__ W, _Float16* __restrict__ P,
                       int N, int KT, int total) {
  const int t = blockIdx.x * blockDim.x + threadIdx.x;
  if (t >= total) return;
  const int e    = t & 15;
  const int lane = (t >> 4) & 31;
  const int frag = t >> 9;
  const int nt = frag / KT;
  const int kt = frag - nt * KT;
  const int k = kt * 32 + ((lane & 16) ? 8 : 0) + (e < 8 ? e : e + 8);
  const int n = nt * 16 + (lane & 15);
  P[t] = (_Float16)W[(size_t)k * N + n];
}

// One thread per (edge, dim). Both aggregation directions + degree counts.
__global__ void scatter_edges(const int* __restrict__ eu, const int* __restrict__ ei,
                              const float* __restrict__ ut, const float* __restrict__ it,
                              float* __restrict__ agg_i, float* __restrict__ agg_u,
                              float* __restrict__ deg_u, float* __restrict__ deg_i,
                              long long total) {
  const long long stride = (long long)gridDim.x * blockDim.x;
  for (long long t = (long long)blockIdx.x * blockDim.x + threadIdx.x; t < total; t += stride) {
    const int e = (int)(t >> 6);
    const int d = (int)(t & 63);
    const int u  = eu[e];
    const int iv = ei[e];
    atomicAdd(&agg_i[(size_t)u  * DIM + d], it[(size_t)iv * DIM + d]);
    atomicAdd(&agg_u[(size_t)iv * DIM + d], ut[(size_t)u  * DIM + d]);
    if (d == 0) {
      atomicAdd(&deg_u[u], 1.0f);
      atomicAdd(&deg_i[iv], 1.0f);
    }
  }
}

// Per-wave LDS layout (bytes):
//   0     : sU   f16[16*64]   (overlaid by sX2 f32[16*64] after GEMM1)
//   2048  : sI   f16[16*64]   (second half of sX2 overlay)
//   4096  : sGU  f16[16*64]   gcn_user_h -> gcn_user_out (in place)
//   6144  : sGI  f16[16*64]   gcn_item_h -> gcn_item_out (in place)
//   8192  : sX1  f16[16*128]
//   12288 : uid int[16], iid int[16], rdu f32[16], rdi f32[16]
#define WAVE_LDS 12544

__global__ __launch_bounds__(128) void fused_mlp(
    const float* __restrict__ user_table, const float* __restrict__ item_table,
    const _Float16* __restrict__ PWu, const float* __restrict__ bu,
    const _Float16* __restrict__ PWi, const float* __restrict__ bi,
    const _Float16* __restrict__ PW1, const float* __restrict__ b1,
    const _Float16* __restrict__ PW2, const float* __restrict__ b2,
    const float* __restrict__ W3, const float* __restrict__ b3,
    const float* __restrict__ user_bias, const float* __restrict__ item_bias,
    const int* __restrict__ user_id, const int* __restrict__ item_id,
    const float* __restrict__ agg_i, const float* __restrict__ agg_u,
    const float* __restrict__ deg_u, const float* __restrict__ deg_i,
    float* __restrict__ out) {
  __shared__ __align__(16) unsigned char smem_raw[4 * WAVE_LDS];
  const int lane = threadIdx.x & 31;
  const int wv   = threadIdx.x >> 5;
  unsigned char* wbase = smem_raw + wv * WAVE_LDS;
  _Float16* sU  = (_Float16*)(wbase);
  _Float16* sI  = (_Float16*)(wbase + 2048);
  _Float16* sGU = (_Float16*)(wbase + 4096);
  _Float16* sGI = (_Float16*)(wbase + 6144);
  _Float16* sX1 = (_Float16*)(wbase + 8192);
  float*    sX2 = (float*)(wbase);          // overlays sU+sI (4 KB) post-GEMM1
  int*      sUid = (int*)(wbase + 12288);
  int*      sIid = (int*)(wbase + 12352);
  float*    sRdu = (float*)(wbase + 12416);
  float*    sRdi = (float*)(wbase + 12480);

  const int tile = blockIdx.x * 4 + wv;   // 1024 tiles of 16 rows
  const int row0 = tile * 16;

  // ---- Stage 1: indices, reciprocal degrees, vectorized gathers (f16) ----
  if (lane < 16) {
    const int u  = user_id[row0 + lane];
    const int iv = item_id[row0 + lane];
    sUid[lane] = u;
    sIid[lane] = iv;
    sRdu[lane] = 1.0f / (deg_u[u] + 1.0f);
    sRdi[lane] = 1.0f / (deg_i[iv] + 1.0f);
  }
  __syncthreads();
  for (int idx = lane; idx < 16 * 16; idx += 32) {   // 8 iters, float4 each
    const int r = idx >> 4, d4 = idx & 15;
    const size_t uo = (size_t)sUid[r] * DIM + d4 * 4;
    const size_t io = (size_t)sIid[r] * DIM + d4 * 4;
    const float4 uu = *(const float4*)(user_table + uo);
    const float4 ii = *(const float4*)(item_table + io);
    const float4 ai = *(const float4*)(agg_i + uo);
    const float4 au = *(const float4*)(agg_u + io);
    const float ru = sRdu[r], ri = sRdi[r];
    const int o = r * 64 + d4 * 4;
    *(v4h*)(sU + o)  = (v4h){(_Float16)uu.x, (_Float16)uu.y, (_Float16)uu.z, (_Float16)uu.w};
    *(v4h*)(sI + o)  = (v4h){(_Float16)ii.x, (_Float16)ii.y, (_Float16)ii.z, (_Float16)ii.w};
    *(v4h*)(sGI + o) = (v4h){(_Float16)(ai.x * ru), (_Float16)(ai.y * ru),
                             (_Float16)(ai.z * ru), (_Float16)(ai.w * ru)}; // gcn_item_h
    *(v4h*)(sGU + o) = (v4h){(_Float16)(au.x * ri), (_Float16)(au.y * ri),
                             (_Float16)(au.z * ri), (_Float16)(au.w * ri)}; // gcn_user_h
  }
  __syncthreads();

  // ---- Stage 2: GCN layers (relu(h @ W + b)), in place ----
  gcn_layer(sGU, PWu, bu, lane);   // gcn_user_out
  gcn_layer(sGI, PWi, bi, lane);   // gcn_item_out
  __syncthreads();

  const int coln  = lane & 15;
  const int rbase = (lane & 16) ? 8 : 0;

  // ---- Stage 3: x = [u*i, u*gi, gu*i, gu*gi]; x1 = tanh(x @ W1 + b1) ----
  // A fragments built as packed products (each 32-wide K-slice stays inside
  // one 64-wide quadrant), preloaded once and reused over 8 n-tiles.
  const _Float16* qx[4] = { sU, sU, sGU, sGU };
  const _Float16* qy[4] = { sI, sGI, sI, sGI };
  v16h xa[8];
#pragma unroll
  for (int kt = 0; kt < 8; ++kt)
    xa[kt] = load_a_prod(qx[kt >> 1], qy[kt >> 1], (kt & 1) * 32, lane);
#pragma unroll
  for (int nt = 0; nt < 8; ++nt) {
    v8f c = {};
#pragma unroll
    for (int kt = 0; kt < 8; ++kt)
      c = wmma16(xa[kt], load_b_packed(PW1, nt * 8 + kt, lane), c);
    const int col = nt * 16 + coln;
    const float bv = b1[col];
#pragma unroll
    for (int j = 0; j < 8; ++j)
      sX1[(rbase + j) * 128 + col] = (_Float16)tanhf(c[j] + bv);
  }
  __syncthreads();

  // ---- Stage 4: x2 = tanh(x1 @ W2 + b2), f32 into sX2 (overlay) ----
  v16h ya[4];
#pragma unroll
  for (int kt = 0; kt < 4; ++kt)
    ya[kt] = load_a_lds(sX1, 128, kt * 32, lane);
#pragma unroll
  for (int nt = 0; nt < 4; ++nt) {
    v8f c = {};
#pragma unroll
    for (int kt = 0; kt < 4; ++kt)
      c = wmma16(ya[kt], load_b_packed(PW2, nt * 4 + kt, lane), c);
    const int col = nt * 16 + coln;
    const float bv = b2[col];
#pragma unroll
    for (int j = 0; j < 8; ++j)
      sX2[(rbase + j) * 64 + col] = tanhf(c[j] + bv);
  }
  __syncthreads();

  // ---- Stage 5: x3 = x2 @ W3 + b3 + user_bias + item_bias ----
  if (lane < 16) {
    float s = 0.f;
#pragma unroll
    for (int d = 0; d < 64; ++d) s += sX2[lane * 64 + d] * W3[d];
    s += b3[0] + user_bias[sUid[lane]] + item_bias[sIid[lane]];
    out[row0 + lane] = s;
  }
}

// ---------------------------------------------------------------------------
// Launch
// ---------------------------------------------------------------------------
extern "C" void kernel_launch(void* const* d_in, const int* in_sizes, int n_in,
                              void* d_out, int out_size, void* d_ws, size_t ws_size,
                              hipStream_t stream) {
  (void)in_sizes; (void)n_in; (void)out_size; (void)ws_size;
  const float* user_table = (const float*)d_in[0];
  const float* item_table = (const float*)d_in[1];
  const float* Wu = (const float*)d_in[2];
  const float* bu = (const float*)d_in[3];
  const float* Wi = (const float*)d_in[4];
  const float* bi = (const float*)d_in[5];
  const float* W1 = (const float*)d_in[6];
  const float* b1 = (const float*)d_in[7];
  const float* W2 = (const float*)d_in[8];
  const float* b2 = (const float*)d_in[9];
  const float* W3 = (const float*)d_in[10];
  const float* b3 = (const float*)d_in[11];
  const float* user_bias = (const float*)d_in[12];
  const float* item_bias = (const float*)d_in[13];
  const int* user_id   = (const int*)d_in[14];
  const int* item_id   = (const int*)d_in[15];
  const int* edge_user = (const int*)d_in[16];
  const int* edge_item = (const int*)d_in[17];
  float* out = (float*)d_out;

  unsigned char* ws = (unsigned char*)d_ws;
  float* agg_i = (float*)(ws);                       // 100000*64 f32 = 25,600,000 B
  float* agg_u = (float*)(ws + 25600000);            //  50000*64 f32 = 12,800,000 B
  float* deg_u = (float*)(ws + 38400000);            // 100000 f32
  float* deg_i = (float*)(ws + 38800000);            //  50000 f32
  _Float16* PWu = (_Float16*)(ws + 39000000);        // 4096 f16 (fragment-major)
  _Float16* PWi = (_Float16*)(ws + 39008192);        // 4096 f16
  _Float16* PW1 = (_Float16*)(ws + 39016384);        // 32768 f16
  _Float16* PW2 = (_Float16*)(ws + 39081920);        // 8192 f16

  // 1) zero accumulators (39,000,000 B = 2,437,500 float4)
  zero_ws<<<2048, 256, 0, stream>>>((float4*)ws, 2437500);

  // 2) weight conversion + fragment packing (f32 [K][N] -> f16 frag-major)
  pack_w<<<16, 256, 0, stream>>>(Wu, PWu, 64, 2, 4096);     // K=64,  N=64,  KT=2
  pack_w<<<16, 256, 0, stream>>>(Wi, PWi, 64, 2, 4096);
  pack_w<<<128, 256, 0, stream>>>(W1, PW1, 128, 8, 32768);  // K=256, N=128, KT=8
  pack_w<<<32, 256, 0, stream>>>(W2, PW2, 64, 4, 8192);     // K=128, N=64,  KT=4

  // 3) edge aggregation: 2M edges x 64 dims
  scatter_edges<<<8192, 256, 0, stream>>>(edge_user, edge_item, user_table, item_table,
                                          agg_i, agg_u, deg_u, deg_i,
                                          (long long)NEDGE * 64);

  // 4) fused gather + GCN + MLP, one wave per 16-row tile (1024 tiles)
  fused_mlp<<<BATCH / 16 / 4, 128, 0, stream>>>(
      user_table, item_table, PWu, bu, PWi, bi, PW1, b1, PW2, b2, W3, b3,
      user_bias, item_bias, user_id, item_id, agg_i, agg_u, deg_u, deg_i, out);
}
